// Attention_67929202753780
// MI455X (gfx1250) — matrix operand order repaired and checked by reading, not compiled, not run on hardware
//
#include <hip/hip_runtime.h>

// ---------------- types ----------------
typedef __attribute__((ext_vector_type(16))) __bf16 v16bf;
typedef __attribute__((ext_vector_type(8)))  __bf16 v8bf;
typedef __attribute__((ext_vector_type(8)))  float  v8f;

#define BN   4
#define NN   3072
#define DIMC 768
#define HH   3
#define HDC  256
#define DD   262      // HD + 6
#define DP   272      // DD padded to mult of 16
#define JREAL 786     // 3*262
#define KPF  800      // JREAL padded to mult of 32
#define ZH   (BN*HH)  // 12

// ---------------- WMMA core ----------------
static __device__ __forceinline__ v8f wmma_bf16(v16bf a, v16bf b, v8f c) {
  // D = A(16x32) * B(32x16) + C, fp32 accumulate
  return __builtin_amdgcn_wmma_f32_16x16x32_bf16(false, a, false, b, (short)0, c, false, false);
}

// A fragment: 16x32 bf16, row-major A[M,K].
// lane<16 : M=lane,   K in {0..7} u {16..23}
// lane>=16: M=lane-16,K in {8..15} u {24..31}
static __device__ __forceinline__ v16bf frag_a(const __bf16* A, long lda, long m0, int k, int lane) {
  const __bf16* p = A + (m0 + (long)(lane & 15)) * lda + k + (lane >> 4) * 8;
  v8bf lo = *(const v8bf*)p;
  v8bf hi = *(const v8bf*)(p + 16);
  return __builtin_shufflevector(lo, hi, 0,1,2,3,4,5,6,7,8,9,10,11,12,13,14,15);
}

// B fragment from Bt[N,K] (i.e. B^T row-major => K contiguous per output column)
// lane<16 : N=lane,    K in {0..7} u {16..23}
// lane>=16: N=lane-16, K in {8..15} u {24..31}
static __device__ __forceinline__ v16bf frag_b(const __bf16* Bt, long ldb, long n0, int k, int lane) {
  const __bf16* p = Bt + (n0 + (long)(lane & 15)) * ldb + k + (lane >> 4) * 8;
  v8bf lo = *(const v8bf*)p;
  v8bf hi = *(const v8bf*)(p + 16);
  return __builtin_shufflevector(lo, hi, 0,1,2,3,4,5,6,7,8,9,10,11,12,13,14,15);
}

// ---------------- generic GEMM with epilogues ----------------
// Wave tile: 32 (M) x 64 (N) = 2x4 accumulators, 8 WMMAs per 6 fragment loads.
// EP: 0=qkv split, 1=attn_f^T (PT), 2=plain bf16 store, 3=fund scatter, 4=fp32 out + bias
struct GArgs {
  const __bf16* A;  const __bf16* Bt;
  long ldA, ldBt, sAz, sBz;
  int  M, Nout, K;
  float scale;
  __bf16 *o_q, *o_k, *o_v;                              // EP0
  const float *rmax, *rsum, *cmax, *csum;               // EP1
  __bf16* OutB; long ldOut, sOz;                        // EP1/2/3
  float* OutF;  const float* bias;                      // EP4
};

template <int EP>
__global__ __launch_bounds__(256) void gemm_ep(GArgs g) {
  const int lane = threadIdx.x & 31;
  const int wv   = threadIdx.x >> 5;
  const int z    = blockIdx.z;
  const int Mgrp = (g.M + 31) >> 5;
  const int Ngrp = (g.Nout + 63) >> 6;
  long wid = (long)blockIdx.x * 8 + wv;
  if (wid >= (long)Mgrp * Ngrp) return;
  const int mg = (int)(wid / Ngrp);
  const int ng = (int)(wid % Ngrp);
  const long m0 = (long)mg * 32;
  const __bf16* A  = g.A  + (long)z * g.sAz;
  const __bf16* Bt = g.Bt + (long)z * g.sBz;

  v8f acc[2][4];
#pragma unroll
  for (int u = 0; u < 2; ++u)
#pragma unroll
    for (int t = 0; t < 4; ++t) acc[u][t] = v8f{};

  for (int k = 0; k < g.K; k += 32) {
    v16bf a0 = frag_a(A, g.ldA, m0, k, lane);
    v16bf a1 = frag_a(A, g.ldA, m0 + 16, k, lane);
#pragma unroll
    for (int t = 0; t < 4; ++t) {
      long n0 = (long)ng * 64 + t * 16;
      if (n0 >= g.Nout) continue;
      v16bf b = frag_b(Bt, g.ldBt, n0, k, lane);
      // prefetch the streaming operand one K-iteration ahead (global_prefetch_b8)
      __builtin_prefetch(Bt + (n0 + (long)(lane & 15)) * g.ldBt + k + 32, 0, 1);
      acc[0][t] = wmma_bf16(a0, b, acc[0][t]);
      acc[1][t] = wmma_bf16(a1, b, acc[1][t]);
    }
  }

  const int lcol = lane & 15;
  const int half = lane >> 4;
#pragma unroll
  for (int t = 0; t < 4; ++t) {
    long n0 = (long)ng * 64 + t * 16;
    if (n0 >= g.Nout) continue;
    const long Cn = n0 + lcol;

    float col_rm = 0.f, col_rinv = 0.f;
    if (EP == 1) {
      long zb = (long)z * NN;
      col_rm   = g.rmax[zb + Cn];
      col_rinv = 1.0f / g.rsum[zb + Cn];
    }
#pragma unroll
    for (int u = 0; u < 2; ++u) {
#pragma unroll
      for (int r = 0; r < 8; ++r) {
        float val = acc[u][t][r];
        const long Rm = m0 + u * 16 + r + half * 8;
        if (Rm >= g.M) continue;  // ragged-M guard (M may not be mult of 32)
        if (EP == 0) {
          long b = Rm / NN, n = Rm % NN;
          int  j = (int)Cn;
          int  s = j / DIMC; int rem = j % DIMC; int h = rem >> 8; int hd = rem & 255;
          __bf16* dst = (s == 0) ? g.o_q : ((s == 1) ? g.o_k : g.o_v);
          float ov = (s == 0) ? val * g.scale : val;
          dst[(((b * HH) + h) * (long)NN + n) * HDC + hd] = (__bf16)ov;
        } else if (EP == 1) {
          long zb = (long)z * NN;
          float cm = g.cmax[zb + Rm];
          float ci = 1.0f / g.csum[zb + Rm];
          float w = __expf(val - col_rm) * __expf(val - cm) * col_rinv * ci;
          g.OutB[(long)z * g.sOz + Rm * g.ldOut + Cn] = (__bf16)w;
        } else if (EP == 2) {
          g.OutB[(long)z * g.sOz + Rm * g.ldOut + Cn] = (__bf16)val;
        } else if (EP == 3) {
          if (Rm < DD && Cn < DD) {
            long b = z / HH, h = z % HH;
            g.OutB[b * (long)(DP * KPF) + Cn * KPF + h * DD + Rm] = (__bf16)val;
          }
        } else { // EP == 4
          if (Rm < DD) {
            g.OutF[((long)z * DD + Rm) * DIMC + Cn] = val + g.bias[Cn];
          }
        }
      }
    }
  }
}

// ---------------- online row max/sum of S = A @ Bt^T ----------------
__global__ __launch_bounds__(256) void stats_kernel(const __bf16* Aall, const __bf16* Btall,
                                                    float* omax, float* osum) {
  const int lane = threadIdx.x & 31;
  const int wv   = threadIdx.x >> 5;
  const int z    = blockIdx.z;
  const __bf16* A  = Aall  + (long)z * NN * HDC;
  const __bf16* Bt = Btall + (long)z * NN * HDC;
  const long m0 = ((long)blockIdx.x * 8 + wv) * 16;

  float runmax[8], runsum[8];
#pragma unroll
  for (int r = 0; r < 8; ++r) { runmax[r] = -3.0e38f; runsum[r] = 0.f; }

  for (int n0 = 0; n0 < NN; n0 += 16) {
    v8f acc = {};
#pragma unroll
    for (int k = 0; k < HDC; k += 32) {
      v16bf a = frag_a(A, HDC, m0, k, lane);
      v16bf b = frag_b(Bt, HDC, n0, k, lane);
      acc = wmma_bf16(a, b, acc);
    }
#pragma unroll
    for (int r = 0; r < 8; ++r) {
      float v = acc[r];
      float mx = v;
      mx = fmaxf(mx, __shfl_xor(mx, 1, 32));
      mx = fmaxf(mx, __shfl_xor(mx, 2, 32));
      mx = fmaxf(mx, __shfl_xor(mx, 4, 32));
      mx = fmaxf(mx, __shfl_xor(mx, 8, 32));
      float nm = fmaxf(runmax[r], mx);
      float e = __expf(v - nm);
      e += __shfl_xor(e, 1, 32);
      e += __shfl_xor(e, 2, 32);
      e += __shfl_xor(e, 4, 32);
      e += __shfl_xor(e, 8, 32);
      runsum[r] = runsum[r] * __expf(runmax[r] - nm) + e;
      runmax[r] = nm;
    }
  }
  if ((lane & 15) == 0) {
    int half = lane >> 4;
    long zb = (long)z * NN;
#pragma unroll
    for (int r = 0; r < 8; ++r) {
      long row = m0 + r + half * 8;
      omax[zb + row] = runmax[r];
      osum[zb + row] = runsum[r];
    }
  }
}

// ---------------- small helper kernels ----------------
__global__ void cvt_f32_bf16(const float* in, __bf16* out, long n) {
  long i = (long)blockIdx.x * blockDim.x + threadIdx.x;
  if (i < n) out[i] = (__bf16)in[i];
}

__global__ void conv_wpf(const float* W, __bf16* out) { // [768][800], zero pad
  long i = (long)blockIdx.x * blockDim.x + threadIdx.x;
  if (i < (long)DIMC * KPF) {
    long c = i / KPF, j = i % KPF;
    out[i] = (j < JREAL) ? (__bf16)W[c * JREAL + j] : (__bf16)0.0f;
  }
}

__global__ void zero_bf16(__bf16* p, long n) {
  long i = (long)blockIdx.x * blockDim.x + threadIdx.x;
  if (i < n) p[i] = (__bf16)0.0f;
}

// vct[z][d][n]: transpose of v + positional rows + zero pad rows
__global__ void build_vct(const __bf16* vh, __bf16* vct) {
  long i = (long)blockIdx.x * blockDim.x + threadIdx.x;
  if (i >= (long)ZH * DP * NN) return;
  long z = i / ((long)DP * NN);
  long rem = i % ((long)DP * NN);
  long d = rem / NN, n = rem % NN;
  float val;
  if (d < HDC) {
    val = (float)vh[(z * NN + n) * HDC + d];
  } else if (d < DD) {
    int row = (int)(n % 48), col = (int)(n / 48);
    float y = -1.0f + 2.0f * row / 47.0f;   // p3 = ys[i % 48]
    float x = -1.0f + 2.0f * col / 63.0f;   // p4 = xs[i / 48]
    int f = (int)(d - HDC);
    val = (f == 0) ? y * y : (f == 1) ? x * x : (f == 2) ? x * y
        : (f == 3) ? y     : (f == 4) ? x     : 1.0f;
  } else {
    val = 0.0f;
  }
  vct[i] = (__bf16)val;
}

// ---------------- host launch ----------------
extern "C" void kernel_launch(void* const* d_in, const int* in_sizes, int n_in,
                              void* d_out, int out_size, void* d_ws, size_t ws_size,
                              hipStream_t stream) {
  const float* x    = (const float*)d_in[0];
  const float* Wqkv = (const float*)d_in[1];
  const float* Wpf  = (const float*)d_in[2];
  const float* bpf  = (const float*)d_in[3];
  float* out = (float*)d_out;
  (void)in_sizes; (void)n_in; (void)out_size; (void)ws_size;

  char* ws = (char*)d_ws;
  size_t off = 0;
  auto take = [&](size_t bytes) -> char* {
    char* p = ws + off;
    off = (off + bytes + 255) & ~(size_t)255;
    return p;
  };
  __bf16* xb    = (__bf16*)take((size_t)BN * NN * DIMC * 2);
  __bf16* wqkvb = (__bf16*)take((size_t)3 * DIMC * DIMC * 2);
  __bf16* wpfb  = (__bf16*)take((size_t)DIMC * KPF * 2);
  __bf16* qh    = (__bf16*)take((size_t)ZH * NN * HDC * 2);
  __bf16* kh    = (__bf16*)take((size_t)ZH * NN * HDC * 2);
  __bf16* vh    = (__bf16*)take((size_t)ZH * NN * HDC * 2);
  __bf16* vct   = (__bf16*)take((size_t)ZH * DP * NN * 2);
  __bf16* f1    = (__bf16*)take((size_t)ZH * DP * NN * 2);
  __bf16* fund5 = (__bf16*)take((size_t)BN * DP * KPF * 2);
  float*  rmax  = (float*)take((size_t)ZH * NN * 4);
  float*  rsum  = (float*)take((size_t)ZH * NN * 4);
  float*  cmax  = (float*)take((size_t)ZH * NN * 4);
  float*  csum  = (float*)take((size_t)ZH * NN * 4);
  __bf16* PT    = (__bf16*)take((size_t)ZH * NN * NN * 2);

  const int T = 256;
  auto blks = [](long n) { return (unsigned)((n + 255) / 256); };
  auto gblk = [](long Mgrp, long Ngrp) { return (unsigned)((Mgrp * Ngrp + 7) / 8); };

  // 1) convert inputs to bf16 (+ padded W_pf)
  cvt_f32_bf16<<<blks((long)BN * NN * DIMC), T, 0, stream>>>(x, xb, (long)BN * NN * DIMC);
  cvt_f32_bf16<<<blks((long)3 * DIMC * DIMC), T, 0, stream>>>(Wqkv, wqkvb, (long)3 * DIMC * DIMC);
  conv_wpf<<<blks((long)DIMC * KPF), T, 0, stream>>>(Wpf, wpfb);
  zero_bf16<<<blks((long)BN * DP * KPF), T, 0, stream>>>(fund5, (long)BN * DP * KPF);

  GArgs g{};
  // 2) qkv GEMM: [12288 x 2304] = xb[12288,768] @ wqkvb^T
  g.A = xb; g.Bt = wqkvb; g.ldA = DIMC; g.ldBt = DIMC; g.sAz = 0; g.sBz = 0;
  g.M = BN * NN; g.Nout = 3 * DIMC; g.K = DIMC;
  g.scale = 0.0625f; // HD^-0.5
  g.o_q = qh; g.o_k = kh; g.o_v = vh;
  gemm_ep<0><<<dim3(gblk(BN * NN / 32, (3 * DIMC) / 64), 1, 1), T, 0, stream>>>(g);

  // 3) v_cat^T with positional rows
  build_vct<<<blks((long)ZH * DP * NN), T, 0, stream>>>(vh, vct);

  // 4) row stats (softmax axis=-1) and col stats (axis=-2)
  stats_kernel<<<dim3((NN / 16) / 8, 1, ZH), T, 0, stream>>>(qh, kh, rmax, rsum);
  stats_kernel<<<dim3((NN / 16) / 8, 1, ZH), T, 0, stream>>>(kh, qh, cmax, csum);

  // 5) PT[m][n] = attn_f[n][m], computed as S^T tiles (A=k, Bt=q)
  g = GArgs{};
  g.A = kh; g.Bt = qh; g.ldA = HDC; g.ldBt = HDC;
  g.sAz = (long)NN * HDC; g.sBz = (long)NN * HDC;
  g.M = NN; g.Nout = NN; g.K = HDC;
  g.rmax = rmax; g.rsum = rsum; g.cmax = cmax; g.csum = csum;
  g.OutB = PT; g.ldOut = NN; g.sOz = (long)NN * NN;
  gemm_ep<1><<<dim3(gblk(NN / 32, NN / 64), 1, ZH), T, 0, stream>>>(g);

  // 6) f1[d][m] = sum_n vct[d][n] * PT[m][n]
  g = GArgs{};
  g.A = vct; g.Bt = PT; g.ldA = NN; g.ldBt = NN;
  g.sAz = (long)DP * NN; g.sBz = (long)NN * NN;
  g.M = DP; g.Nout = NN; g.K = NN;
  g.OutB = f1; g.ldOut = NN; g.sOz = (long)DP * NN;
  gemm_ep<2><<<dim3(gblk((DP + 31) / 32, NN / 64), 1, ZH), T, 0, stream>>>(g);

  // 7) fund[d][e] = sum_m f1[d][m] * vct[e][m]  -> scatter to fund5[b][e][h*262+d]
  g = GArgs{};
  g.A = f1; g.Bt = vct; g.ldA = NN; g.ldBt = NN;
  g.sAz = (long)DP * NN; g.sBz = (long)DP * NN;
  g.M = DP; g.Nout = DP; g.K = NN;
  g.OutB = fund5;
  gemm_ep<3><<<dim3(gblk((DP + 31) / 32, (DP + 63) / 64), 1, ZH), T, 0, stream>>>(g);

  // 8) out[b][e][c] = sum_j fund5[b][e][j] * wpfb[c][j] + b_pf[c]
  g = GArgs{};
  g.A = fund5; g.Bt = wpfb; g.ldA = KPF; g.ldBt = KPF;
  g.sAz = (long)DP * KPF; g.sBz = 0;
  g.M = DP; g.Nout = DIMC; g.K = KPF;
  g.OutF = out; g.bias = bpf;
  gemm_ep<4><<<dim3(gblk((DP + 31) / 32, DIMC / 64), 1, BN), T, 0, stream>>>(g);
}
